// AttentionSparseMax_12103217840497
// MI455X (gfx1250) — matrix operand order repaired and checked by reading, not compiled, not run on hardware
//
#include <hip/hip_runtime.h>
#include <hip/hip_bf16.h>

// ---------------------------------------------------------------------------
// CDNA5 (gfx1250) fused sparsemax-attention + MLP, bf16 WMMA everywhere.
// bf16 A operands are 2x16B per-lane loads, B operands 16 strided loads (weights pre-transposed,
// K-projection stored transposed); B elements are read down K with row stride ld.
// ---------------------------------------------------------------------------

typedef __bf16 bf16_t;
typedef __attribute__((ext_vector_type(16))) __bf16 v16bf;
typedef __attribute__((ext_vector_type(8)))  float  v8f;

__device__ __forceinline__ v8f wmma_bf16(v16bf a, v16bf b, v8f c) {
    // D = A(16x32 bf16) * B(32x16 bf16) + C(16x16 f32)
    return __builtin_amdgcn_wmma_f32_16x16x32_bf16(
        false, a, false, b, (short)0, c, false, false);
}

// A-operand (16x32 bf16): lane(0-15)=row m; halves pick K {0..7,16..23} vs
// {8..15,24..31}. p points at row + k0 + 8*half; elems 0-7 at p, 8-15 at p+16.
__device__ __forceinline__ v16bf ldA16(const bf16_t* __restrict__ p) {
    v16bf a;
    *(uint4*)&a       = *(const uint4*)p;
    *(((uint4*)&a)+1) = *(const uint4*)(p + 16);
    return a;
}

// B-operand (32x16 bf16): lane%16 = N column; p at K row 8*half; elems 0-7 rows
// +0..7, elems 8-15 rows +16..23 (row stride ld).
__device__ __forceinline__ v16bf ldB16(const bf16_t* __restrict__ p, int ld) {
    v16bf b;
#pragma unroll
    for (int s = 0; s < 16; ++s) b[s] = p[(size_t)(s < 8 ? s : s + 8) * ld];
    return b;
}

// ---------------------------------------------------------------------------
// f32 -> bf16 row-wise conversion (optionally strided dst, for concat).
// ---------------------------------------------------------------------------
__global__ void cvt_rows_k(const float* __restrict__ src, bf16_t* __restrict__ dst,
                           int cols, int dld, int coff) {
    const int r = blockIdx.y;
    const int c = blockIdx.x * blockDim.x + threadIdx.x;
    if (c < cols)
        dst[(size_t)r * dld + coff + c] = (bf16_t)src[(size_t)r * cols + c];
}

// f32 [R,C] row-major -> bf16 transposed [C, ldt] (dst[c*ldt + r] = src[r*C + c]).
// Tiled through LDS so both sides stay coalesced. One-time cost per weight.
__global__ void transpose_bf16_k(const float* __restrict__ src, bf16_t* __restrict__ dst,
                                 int R, int C, int ldt) {
    __shared__ float tile[32][33];
    const int c0 = blockIdx.x * 32, r0 = blockIdx.y * 32;
    const int tx = threadIdx.x, ty = threadIdx.y;      // block (32, 8)
    for (int i = ty; i < 32; i += 8) {
        const int r = r0 + i, c = c0 + tx;
        tile[i][tx] = (r < R && c < C) ? src[(size_t)r * C + c] : 0.0f;
    }
    __syncthreads();
    for (int i = ty; i < 32; i += 8) {
        const int c = c0 + i, r = r0 + tx;
        if (c < C && r < R)
            dst[(size_t)c * ldt + r] = (bf16_t)tile[tx][i];
    }
}

// ---------------------------------------------------------------------------
// Generic bf16 WMMA GEMM: C[M,N] = A[M,K] * W^T + bias, W given pre-transposed
// as BT[K, ldbt>=N]. Optional ReLU; f32 and/or bf16 output; bf16 output can be
// stored transposed (for the attention K projection). Wave = 32x32 C tile.
// ---------------------------------------------------------------------------
__global__ void __launch_bounds__(256)
gemm_wmma_k(const bf16_t* __restrict__ A, int lda,
            const bf16_t* __restrict__ BT, int ldbt,
            const float* __restrict__ bias,
            float* __restrict__ Cf, int ldcf,
            bf16_t* __restrict__ Cb, int ldcb, int coff, int cbT,
            int M, int N, int K, int relu, int tilesN) {
    const int wid  = blockIdx.x * (blockDim.x >> 5) + (threadIdx.x >> 5);
    const int tm   = wid / tilesN;
    const int tn   = wid - tm * tilesN;
    const int r0   = tm * 32, c0 = tn * 32;
    if (r0 >= M) return;
    const int lane = threadIdx.x & 31;
    const int half = lane >> 4, m = lane & 15;

    const bf16_t* pa = A  + (size_t)(r0 + m) * lda + 8 * half;
    const bf16_t* pb = BT + (size_t)(8 * half) * ldbt + c0 + m;
    const size_t a1off = (size_t)16 * lda;
    const size_t pbstep = (size_t)32 * ldbt;

    v8f c00 = {}, c01 = {}, c10 = {}, c11 = {};
    for (int k0 = 0; k0 < K; k0 += 32) {
        v16bf a0 = ldA16(pa);
        v16bf a1 = ldA16(pa + a1off);
        v16bf b0 = ldB16(pb, ldbt);
        v16bf b1 = ldB16(pb + 16, ldbt);
        c00 = wmma_bf16(a0, b0, c00);
        c01 = wmma_bf16(a0, b1, c01);
        c10 = wmma_bf16(a1, b0, c10);
        c11 = wmma_bf16(a1, b1, c11);
        pa += 32;
        pb += pbstep;
    }

    v8f accs[4] = {c00, c01, c10, c11};
    const int rbs[4] = {r0, r0, r0 + 16, r0 + 16};
    const int cbs[4] = {c0, c0 + 16, c0, c0 + 16};
#pragma unroll
    for (int t = 0; t < 4; ++t) {
#pragma unroll
        for (int r = 0; r < 8; ++r) {
            const int row = rbs[t] + r + 8 * half;
            const int col = cbs[t] + m;
            if (col < N) {
                float v = accs[t][r] + (bias ? bias[col] : 0.0f);
                if (relu) v = fmaxf(v, 0.0f);
                if (Cf) Cf[(size_t)row * ldcf + col] = v;
                if (Cb) {
                    if (cbT) Cb[(size_t)(coff + col) * ldcb + row] = (bf16_t)v;
                    else     Cb[(size_t)row * ldcb + coff + col]  = (bf16_t)v;
                }
            }
        }
    }
}

// ---------------------------------------------------------------------------
// Fused sparsemax attention. One workgroup = one head x one 16-query tile.
// LDS: S[16][M] f32 (256KB) + tau[16] + Opart[16][64]  -> fits 320KB WGP LDS.
// Phase 1: S = (Q_h K_h^T)/sqrt(d) via WMMA (K given transposed: kT[d, M]).
// Phase 2: per-row tau by bisection on sum(relu(z - tau)) = 1.
// Phase 3: O = relu(S - tau) @ V_h via WMMA, A formed on the fly from LDS.
// ---------------------------------------------------------------------------
#define DH 64

__global__ void __launch_bounds__(256)
attn_sparsemax_k(const bf16_t* __restrict__ Q,
                 const bf16_t* __restrict__ kT,   // [d, Mm]
                 const bf16_t* __restrict__ Vb,   // [Mm, d]
                 bf16_t* __restrict__ Out,        // [Nq, d]
                 int Mm, int d, int H, float scale) {
    extern __shared__ char smem_raw[];
    float* S      = (float*)smem_raw;                 // 16 * Mm floats
    float* tauLds = S + (size_t)16 * Mm;              // 16 floats
    float* Opart  = tauLds + 16;                      // 16 * 64 floats

    const int h    = blockIdx.x % H;
    const int qt   = blockIdx.x / H;
    const int row0 = qt * 16;
    const int hoff = h * DH;
    const int wave = threadIdx.x >> 5;
    const int lane = threadIdx.x & 31;
    const int half = lane >> 4, m = lane & 15;

    // ---- Phase 1: scores into LDS ----
    const bf16_t* pq = Q + (size_t)(row0 + m) * d + hoff + 8 * half;
    v16bf aq0 = ldA16(pq);                 // dh K-range 0..31
    v16bf aq1 = ldA16(pq + 32);            // dh K-range 32..63
    const bf16_t* pk0 = kT + (size_t)(hoff + 8 * half) * Mm + m;
    const bf16_t* pk1 = pk0 + (size_t)32 * Mm;
    for (int jt = wave; jt < Mm / 16; jt += 8) {
        const int m0 = jt * 16;
        v16bf b0 = ldB16(pk0 + m0, Mm);
        v16bf b1 = ldB16(pk1 + m0, Mm);
        v8f c = {};
        c = wmma_bf16(aq0, b0, c);
        c = wmma_bf16(aq1, b1, c);
#pragma unroll
        for (int r = 0; r < 8; ++r)
            S[(size_t)(r + 8 * half) * Mm + m0 + m] = c[r] * scale;
    }
    __syncthreads();

    // ---- Phase 2: sparsemax threshold per row (wave handles rows w, w+8) ----
#pragma unroll
    for (int rr = 0; rr < 2; ++rr) {
        const int r = wave + 8 * rr;
        const float* row = S + (size_t)r * Mm;
        float mx = -1e30f;
        for (int c0 = lane; c0 < Mm; c0 += 32) mx = fmaxf(mx, row[c0]);
#pragma unroll
        for (int off = 16; off > 0; off >>= 1)
            mx = fmaxf(mx, __shfl_xor(mx, off, 32));
        float lo = mx - 1.0f, hi = mx;   // sum(tau=lo) >= 1 >= sum(tau=hi)
#pragma unroll 1
        for (int it = 0; it < 30; ++it) {
            const float t = 0.5f * (lo + hi);
            float s = 0.0f;
            for (int c0 = lane; c0 < Mm; c0 += 32)
                s += fmaxf(row[c0] - t, 0.0f);
#pragma unroll
            for (int off = 16; off > 0; off >>= 1)
                s += __shfl_xor(s, off, 32);
            if (s >= 1.0f) lo = t; else hi = t;
        }
        if (lane == 0) tauLds[r] = 0.5f * (lo + hi);
    }
    __syncthreads();

    // ---- Phase 3: O = P @ V_h. 8 waves: 4 column tiles x 2 K-halves ----
    const int nt = wave & 3;       // dh column tile (16 cols each)
    const int kh = wave >> 2;      // K half
    const int n0 = nt * 16;
    const float taum = tauLds[m];  // tau for this lane's A-row
    const float* ps = S + (size_t)m * Mm + 8 * half;   // + k0 per step
    const bf16_t* pv = Vb + (size_t)(8 * half) * d + hoff + n0 + m;
    v8f acc = {};
    const int kbeg = kh * (Mm / 2), kend = kbeg + Mm / 2;
    for (int k0 = kbeg; k0 < kend; k0 += 32) {
        // A = relu(S - tau) -> bf16, built from four float4 LDS loads.
        const float* p = ps + k0;
        float4 f0 = *(const float4*)(p);
        float4 f1 = *(const float4*)(p + 4);
        float4 f2 = *(const float4*)(p + 16);
        float4 f3 = *(const float4*)(p + 20);
        v16bf a;
        a[0]  = (bf16_t)fmaxf(f0.x - taum, 0.0f);
        a[1]  = (bf16_t)fmaxf(f0.y - taum, 0.0f);
        a[2]  = (bf16_t)fmaxf(f0.z - taum, 0.0f);
        a[3]  = (bf16_t)fmaxf(f0.w - taum, 0.0f);
        a[4]  = (bf16_t)fmaxf(f1.x - taum, 0.0f);
        a[5]  = (bf16_t)fmaxf(f1.y - taum, 0.0f);
        a[6]  = (bf16_t)fmaxf(f1.z - taum, 0.0f);
        a[7]  = (bf16_t)fmaxf(f1.w - taum, 0.0f);
        a[8]  = (bf16_t)fmaxf(f2.x - taum, 0.0f);
        a[9]  = (bf16_t)fmaxf(f2.y - taum, 0.0f);
        a[10] = (bf16_t)fmaxf(f2.z - taum, 0.0f);
        a[11] = (bf16_t)fmaxf(f2.w - taum, 0.0f);
        a[12] = (bf16_t)fmaxf(f3.x - taum, 0.0f);
        a[13] = (bf16_t)fmaxf(f3.y - taum, 0.0f);
        a[14] = (bf16_t)fmaxf(f3.z - taum, 0.0f);
        a[15] = (bf16_t)fmaxf(f3.w - taum, 0.0f);
        v16bf b = ldB16(pv + (size_t)k0 * d, d);
        acc = wmma_bf16(a, b, acc);
    }
    if (kh == 1) {
#pragma unroll
        for (int r = 0; r < 8; ++r)
            Opart[(r + 8 * half) * DH + n0 + m] = acc[r];
    }
    __syncthreads();
    if (kh == 0) {
#pragma unroll
        for (int r = 0; r < 8; ++r) {
            const float v = acc[r] + Opart[(r + 8 * half) * DH + n0 + m];
            Out[(size_t)(row0 + r + 8 * half) * d + hoff + n0 + m] = (bf16_t)v;
        }
    }
}

// ---------------------------------------------------------------------------
// Host orchestration
// ---------------------------------------------------------------------------
static inline int gemm_blocks(int M, int N) {
    const int tilesM = (M + 31) / 32, tilesN = (N + 31) / 32;
    return (tilesM * tilesN + 7) / 8;
}

extern "C" void kernel_launch(void* const* d_in, const int* in_sizes, int n_in,
                              void* d_out, int out_size, void* d_ws, size_t ws_size,
                              hipStream_t stream) {
    (void)in_sizes; (void)n_in; (void)out_size; (void)ws_size;
    const int N = 2048, Mm = 4096, d = 1024, H = 16, OUTD = 1000, OUTP = 1024;
    const int d2 = 2 * d, d4 = 4 * d;

    const float* enc = (const float*)d_in[0];
    const float* mem = (const float*)d_in[1];
    const float* Wq  = (const float*)d_in[2];  const float* bq = (const float*)d_in[3];
    const float* Wk  = (const float*)d_in[4];  const float* bk = (const float*)d_in[5];
    const float* Wv  = (const float*)d_in[6];  const float* bv = (const float*)d_in[7];
    const float* Wo  = (const float*)d_in[8];  const float* bo = (const float*)d_in[9];
    const float* W1  = (const float*)d_in[10]; const float* b1 = (const float*)d_in[11];
    const float* W2  = (const float*)d_in[12]; const float* b2 = (const float*)d_in[13];
    float* out = (float*)d_out;

    // Workspace carve-up (bf16 buffers), 256B aligned.
    size_t off = 0;
    auto carve = [&](size_t elems) {
        size_t o = off; off += ((elems * sizeof(bf16_t)) + 255) & ~(size_t)255; return o;
    };
    char* ws = (char*)d_ws;
    bf16_t* fin_bf = (bf16_t*)(ws + carve((size_t)N * d2));     // [N,2d]: [enc | proj]
    bf16_t* mem_bf = (bf16_t*)(ws + carve((size_t)Mm * d));
    bf16_t* WqT = (bf16_t*)(ws + carve((size_t)d * d));         // [d,d]   (K x N)
    bf16_t* WkT = (bf16_t*)(ws + carve((size_t)d * d));
    bf16_t* WvT = (bf16_t*)(ws + carve((size_t)d * d));
    bf16_t* WoT = (bf16_t*)(ws + carve((size_t)d * d));
    bf16_t* W1T = (bf16_t*)(ws + carve((size_t)d2 * d4));       // [2d,4d]
    bf16_t* W2T = (bf16_t*)(ws + carve((size_t)d4 * OUTP));     // [4d,1024pad]
    bf16_t* q_bf = (bf16_t*)(ws + carve((size_t)N * d));        // [N,d]
    bf16_t* kT   = (bf16_t*)(ws + carve((size_t)d * Mm));       // [d,Mm] transposed
    bf16_t* v_bf = (bf16_t*)(ws + carve((size_t)Mm * d));       // [Mm,d]
    bf16_t* ao_bf = (bf16_t*)(ws + carve((size_t)N * d));       // attention output
    bf16_t* h_bf  = (bf16_t*)(ws + carve((size_t)N * d4));      // [N,4d]

    // --- conversions / transposes (one-time per call, coalesced) ---
    {
        dim3 b(256, 1);
        cvt_rows_k<<<dim3((d + 255) / 256, N), b, 0, stream>>>(enc, fin_bf, d, d2, 0);
        cvt_rows_k<<<dim3((d + 255) / 256, Mm), b, 0, stream>>>(mem, mem_bf, d, d, 0);
    }
    {
        dim3 b(32, 8);
        transpose_bf16_k<<<dim3(d / 32, d / 32), b, 0, stream>>>(Wq, WqT, d, d, d);
        transpose_bf16_k<<<dim3(d / 32, d / 32), b, 0, stream>>>(Wk, WkT, d, d, d);
        transpose_bf16_k<<<dim3(d / 32, d / 32), b, 0, stream>>>(Wv, WvT, d, d, d);
        transpose_bf16_k<<<dim3(d / 32, d / 32), b, 0, stream>>>(Wo, WoT, d, d, d);
        transpose_bf16_k<<<dim3(d2 / 32, d4 / 32), b, 0, stream>>>(W1, W1T, d4, d2, d4);
        transpose_bf16_k<<<dim3(d4 / 32, (OUTD + 31) / 32), b, 0, stream>>>(W2, W2T, OUTD, d4, OUTP);
    }

    // --- projections ---
    gemm_wmma_k<<<gemm_blocks(N, d), 256, 0, stream>>>(
        fin_bf, d2, WqT, d, bq, nullptr, 0, q_bf, d, 0, 0, N, d, d, 0, (d + 31) / 32);
    gemm_wmma_k<<<gemm_blocks(Mm, d), 256, 0, stream>>>(          // K stored transposed
        mem_bf, d, WkT, d, bk, nullptr, 0, kT, Mm, 0, 1, Mm, d, d, 0, (d + 31) / 32);
    gemm_wmma_k<<<gemm_blocks(Mm, d), 256, 0, stream>>>(
        mem_bf, d, WvT, d, bv, nullptr, 0, v_bf, d, 0, 0, Mm, d, d, 0, (d + 31) / 32);

    // --- fused sparsemax attention: one block per (head, 16-query tile) ---
    const size_t smem = ((size_t)16 * Mm + 16 + 16 * DH) * sizeof(float); // ~260KB
    const float scale = 0.03125f; // 1/sqrt(1024) -- full-d scaling per reference
    attn_sparsemax_k<<<(N / 16) * H, 256, smem, stream>>>(
        q_bf, kT, v_bf, ao_bf, Mm, d, H, scale);

    // --- proj = attn_out @ Wo^T + bo -> right half of fin ---
    gemm_wmma_k<<<gemm_blocks(N, d), 256, 0, stream>>>(
        ao_bf, d, WoT, d, bo, nullptr, 0, fin_bf, d2, d, 0, N, d, d, 0, (d + 31) / 32);

    // --- h = relu(fin @ W1^T + b1) ---
    gemm_wmma_k<<<gemm_blocks(N, d4), 256, 0, stream>>>(
        fin_bf, d2, W1T, d4, b1, nullptr, 0, h_bf, d4, 0, 0, N, d4, d2, 1, (d4 + 31) / 32);

    // --- out = h @ W2^T + b2 (f32, N=1000 edge-masked; W2T padded to 1024) ---
    gemm_wmma_k<<<gemm_blocks(N, OUTD), 256, 0, stream>>>(
        h_bf, d4, W2T, OUTP, b2, out, OUTD, nullptr, 0, 0, 0, N, OUTD, d4, 0, (OUTD + 31) / 32);
}
